// CeATTEnhancedCrossAttention_72550587564731
// MI455X (gfx1250) — matrix operand order repaired and checked by reading, not compiled, not run on hardware
//
#include <hip/hip_runtime.h>
#include <hip/hip_bf16.h>
#include <stdint.h>

typedef __bf16 bf16;
typedef __attribute__((ext_vector_type(8)))  float v8f;
typedef __attribute__((ext_vector_type(8)))  bf16  v8bf;
typedef __attribute__((ext_vector_type(16))) bf16  v16bf;

#define B_   16
#define T_   243
#define J_   17
#define C_   256
#define H_   8
#define HD_  32
#define TP_  256                       // padded T for tiling
#define BHJ_ (B_*H_*J_)                // 2176
#define ROWS_ (B_*T_*J_)               // 66096 (== 4131 * 16 exactly)
#define QKV_ELEMS ((long)BHJ_*TP_*HD_) // 17,825,792 bf16 per projection buffer
#define ATTN_ELEMS ((long)BHJ_*(long)T_*(long)T_)
#define MASKV (-3.0e38f)

static __device__ inline v16bf join8(v8bf l, v8bf h) {
  return __builtin_shufflevector(l, h, 0,1,2,3,4,5,6,7,8,9,10,11,12,13,14,15);
}
static __device__ inline v16bf cvt16(v8f lo, v8f hi) {
  return join8(__builtin_convertvector(lo, v8bf),
               __builtin_convertvector(hi, v8bf));
}
static __device__ inline v16bf ld16bf(const bf16* p) {       // 16B-aligned loads
  return join8(*(const v8bf*)p, *(const v8bf*)(p + 8));
}

// Async 16B-per-lane copy global -> LDS (VFLAT GLOBAL_LOAD_ASYNC_TO_LDS_B128,
// tracked by ASYNCcnt). Inline asm: bypasses the OpenCL-AS-typed clang builtin
// that HIP source cannot name. LDS byte offset = low 32 bits of the generic
// pointer (aperture lives in the high DWORD per the gfx1250 aperture check).
static __device__ inline void async_copy16(const bf16* gsrc, bf16* ldst) {
  uint64_t ga = (uint64_t)(uintptr_t)gsrc;
  uint32_t la = (uint32_t)(uintptr_t)ldst;
  asm volatile("global_load_async_to_lds_b128 %0, %1, off"
               :: "v"(la), "v"(ga) : "memory");
}
static __device__ inline void async_wait0() {
  asm volatile("s_wait_asynccnt 0x0" ::: "memory");
}

// ---------------------------------------------------------------------------
// Kernel 1: Y = X @ W^T, X f32 [ROWS_,256], W f32 [256,256].
// Result scattered (bf16) into [B,H,J,TP_,HD_] for the attention kernel.
// One wave per (16-row tile, 16-col tile); 8 waves per block.
// ---------------------------------------------------------------------------
__global__ __launch_bounds__(256) void proj_qkv(const float* __restrict__ X,
                                                const float* __restrict__ W,
                                                bf16* __restrict__ dst) {
  const int lane = threadIdx.x & 31;
  const int wave = threadIdx.x >> 5;
  const int task = blockIdx.x * 8 + wave;   // 66096 tasks total
  const int rt = task >> 4;                 // row tile (0..4130)
  const int ct = task & 15;                 // col tile (0..15)
  const int hl = lane >> 4;                 // lane half
  const float* xrow = X + (long)(rt*16 + (lane & 15)) * C_;
  const float* wrow = W + (long)(ct*16 + (lane & 15)) * C_;

  v8f acc = {};
  #pragma unroll
  for (int kk = 0; kk < 8; ++kk) {
    const int k0 = kk*32 + hl*16;           // A: 16 contiguous K per lane half
    v16bf a = cvt16(*(const v8f*)(xrow + k0), *(const v8f*)(xrow + k0 + 8));
    const int kb = kk*32 + hl*16;           // B: 16 contiguous K per lane half
    v16bf b = cvt16(*(const v8f*)(wrow + kb), *(const v8f*)(wrow + kb + 8));
    acc = __builtin_amdgcn_wmma_f32_16x16x32_bf16(false, a, false, b,
                                                  (short)0, acc, false, false);
  }
  const int c = ct*16 + (lane & 15);
  const int h = c >> 5, e = c & 31;
  #pragma unroll
  for (int v = 0; v < 8; ++v) {
    const int r = rt*16 + v + 8*hl;         // D layout: M = v + 8*(lane/16)
    const int bb = r / (T_*J_);
    const int rem = r % (T_*J_);
    const int t = rem / J_, j = rem % J_;
    dst[(((long)(bb*H_ + h)*J_ + j)*TP_ + t)*HD_ + e] = (bf16)acc[v];
  }
}

// ---------------------------------------------------------------------------
// Kernel 2: fused scores + softmax + P*V for one (b,h,j) and 64 Tq rows.
// 4 waves / block; each wave owns a 16-row stripe of S = Q K^T (full 256 cols).
// K tile staged with GLOBAL_LOAD_ASYNC_TO_LDS_B128 (ASYNCcnt).
// ---------------------------------------------------------------------------
__global__ __launch_bounds__(128) void attn_fused(const bf16* __restrict__ Qp,
                                                  const bf16* __restrict__ Kp,
                                                  const bf16* __restrict__ Vp,
                                                  float* __restrict__ attn,
                                                  float* __restrict__ O) {
  __shared__ __align__(64) bf16 Ksh[TP_*HD_];        // [t][e]   16 KB
  __shared__ __align__(64) bf16 Vtsh[HD_*TP_];       // [e][t]   16 KB
  __shared__ __align__(64) bf16 Psh[4][16*TP_];      // per-wave P, 32 KB
  const int tid = threadIdx.x, lane = tid & 31, wave = tid >> 5;
  const int bhj  = blockIdx.x >> 2;
  const int mrow0 = (blockIdx.x & 3)*64 + wave*16;
  const int bb = bhj / (H_*J_);
  const int h  = (bhj / J_) % H_;
  const int j  = bhj % J_;
  const long base = (long)bhj * (TP_*HD_);
  const int hl = lane >> 4;

  // stage K (row-major): pure copy -> async DMA to LDS, no VGPR round-trip
  #pragma unroll
  for (int it = 0; it < (TP_*HD_/8)/128; ++it) {
    const int i = tid + it*128;
    async_copy16(Kp + base + (long)i*8, &Ksh[i*8]);
  }

  // stage V transposed into Vtsh (needs reshape -> generic path)
  for (int i = tid; i < TP_*HD_/8; i += 128) {
    const int t = (i*8) >> 5, e0 = (i*8) & 31;
    v8bf vv = *(const v8bf*)&Vp[base + (long)i*8];
    #pragma unroll
    for (int u = 0; u < 8; ++u) Vtsh[(e0+u)*TP_ + t] = vv[u];
  }

  async_wait0();
  __syncthreads();

  // Q fragment: whole hd=32 fits one WMMA K
  const bf16* qrow = Qp + (base + (long)(mrow0 + (lane & 15))*HD_);
  v16bf aQ = join8(*(const v8bf*)(qrow + hl*16), *(const v8bf*)(qrow + hl*16 + 8));

  // S = Q K^T over 16 N-tiles, kept in registers
  const float scale = 0.17677669529663687f;  // 1/sqrt(32)
  v8f sv[16];
  #pragma unroll
  for (int nt = 0; nt < 16; ++nt) {
    v16bf bK = ld16bf(&Ksh[(nt*16 + (lane & 15))*HD_ + hl*16]);
    v8f c = {};
    c = __builtin_amdgcn_wmma_f32_16x16x32_bf16(false, aQ, false, bK,
                                                (short)0, c, false, false);
    const bool tok = (nt*16 + (lane & 15)) < T_;
    #pragma unroll
    for (int v = 0; v < 8; ++v) sv[nt][v] = tok ? c[v]*scale : MASKV;
  }

  // row softmax (rows live across the 16 lanes of each half-wave)
  #pragma unroll
  for (int v = 0; v < 8; ++v) {
    float m = sv[0][v];
    #pragma unroll
    for (int nt = 1; nt < 16; ++nt) m = fmaxf(m, sv[nt][v]);
    #pragma unroll
    for (int off = 1; off < 16; off <<= 1) m = fmaxf(m, __shfl_xor(m, off, 32));
    float s = 0.f;
    #pragma unroll
    for (int nt = 0; nt < 16; ++nt) { float p = __expf(sv[nt][v] - m); sv[nt][v] = p; s += p; }
    #pragma unroll
    for (int off = 1; off < 16; off <<= 1) s += __shfl_xor(s, off, 32);
    const float inv = 1.0f / s;
    #pragma unroll
    for (int nt = 0; nt < 16; ++nt) sv[nt][v] *= inv;
  }

  // write attn (f32, exactly once) and stage P (bf16) for the PV GEMM
  #pragma unroll
  for (int nt = 0; nt < 16; ++nt) {
    const int t = nt*16 + (lane & 15);
    #pragma unroll
    for (int v = 0; v < 8; ++v) {
      const int mloc = v + 8*hl;
      const int m = mrow0 + mloc;
      if (t < T_ && m < T_)
        attn[((long)bhj*T_ + m)*T_ + t] = sv[nt][v];
      Psh[wave][mloc*TP_ + t] = (bf16)sv[nt][v];
    }
  }
  __syncthreads();

  // O = P * V  (M=16, N=32 as two tiles, K=256 in 8 chunks)
  #pragma unroll
  for (int et = 0; et < 2; ++et) {
    v8f acc = {};
    #pragma unroll
    for (int kc = 0; kc < 8; ++kc) {
      const bf16* prow = &Psh[wave][(lane & 15)*TP_ + kc*32 + hl*16];
      v16bf aP = join8(*(const v8bf*)prow, *(const v8bf*)(prow + 8));
      v16bf bV = ld16bf(&Vtsh[(et*16 + (lane & 15))*TP_ + kc*32 + hl*16]);
      acc = __builtin_amdgcn_wmma_f32_16x16x32_bf16(false, aP, false, bV,
                                                    (short)0, acc, false, false);
    }
    const int e = et*16 + (lane & 15);
    #pragma unroll
    for (int v = 0; v < 8; ++v) {
      const int m = mrow0 + v + 8*hl;
      if (m < T_)
        O[(((long)bb*T_ + m)*J_ + j)*C_ + h*HD_ + e] = acc[v];
    }
  }
}

// ---------------------------------------------------------------------------
// Kernel 3: out = O @ Wp^T + bp, f32 in, f32 out
// ---------------------------------------------------------------------------
__global__ __launch_bounds__(256) void proj_out(const float* __restrict__ X,
                                                const float* __restrict__ W,
                                                const float* __restrict__ bias,
                                                float* __restrict__ Y) {
  const int lane = threadIdx.x & 31;
  const int wave = threadIdx.x >> 5;
  const int task = blockIdx.x * 8 + wave;
  const int rt = task >> 4, ct = task & 15;
  const int hl = lane >> 4;
  const float* xrow = X + (long)(rt*16 + (lane & 15)) * C_;
  const float* wrow = W + (long)(ct*16 + (lane & 15)) * C_;

  v8f acc = {};
  #pragma unroll
  for (int kk = 0; kk < 8; ++kk) {
    const int k0 = kk*32 + hl*16;
    v16bf a = cvt16(*(const v8f*)(xrow + k0), *(const v8f*)(xrow + k0 + 8));
    const int kb = kk*32 + hl*16;
    v16bf b = cvt16(*(const v8f*)(wrow + kb), *(const v8f*)(wrow + kb + 8));
    acc = __builtin_amdgcn_wmma_f32_16x16x32_bf16(false, a, false, b,
                                                  (short)0, acc, false, false);
  }
  const float bv = bias[ct*16 + (lane & 15)];
  #pragma unroll
  for (int v = 0; v < 8; ++v)
    Y[(long)(rt*16 + v + 8*hl)*C_ + ct*16 + (lane & 15)] = acc[v] + bv;
}

// ---------------------------------------------------------------------------
extern "C" void kernel_launch(void* const* d_in, const int* in_sizes, int n_in,
                              void* d_out, int out_size, void* d_ws, size_t ws_size,
                              hipStream_t stream) {
  const float* q  = (const float*)d_in[0];
  const float* kv = (const float*)d_in[1];
  const float* Wq = (const float*)d_in[2];
  const float* Wk = (const float*)d_in[3];
  const float* Wv = (const float*)d_in[4];
  const float* Wp = (const float*)d_in[5];
  const float* bp = (const float*)d_in[6];

  // workspace: Qp/Kp/Vp bf16 [BHJ,256,32] (3 x 35.6 MB) + O f32 [B,T,J,C] (67.7 MB)
  bf16* Qp = (bf16*)d_ws;
  bf16* Kp = Qp + QKV_ELEMS;
  bf16* Vp = Kp + QKV_ELEMS;
  float* O = (float*)(Vp + QKV_ELEMS);

  float* attn = (float*)d_out;
  float* out  = attn + ATTN_ELEMS;

  // zero the padded projection buffers (t in [243,256) must be 0)
  (void)hipMemsetAsync(d_ws, 0, (size_t)QKV_ELEMS * 3 * sizeof(bf16), stream);

  const int projBlocks = (ROWS_/16) * 16 / 8;   // 4131 row tiles * 16 col tiles / 8 waves
  proj_qkv<<<projBlocks, 256, 0, stream>>>(q,  Wq, Qp);
  proj_qkv<<<projBlocks, 256, 0, stream>>>(kv, Wk, Kp);
  proj_qkv<<<projBlocks, 256, 0, stream>>>(kv, Wv, Vp);
  attn_fused<<<BHJ_ * 4, 128, 0, stream>>>(Qp, Kp, Vp, attn, O);
  proj_out<<<projBlocks, 256, 0, stream>>>(O, Wp, bp, out);
}
